// OnlineClusteringHead_59828894433941
// MI455X (gfx1250) — matrix operand order repaired and checked by reading, not compiled, not run on hardware
//
#include <hip/hip_runtime.h>

// ---------------------------------------------------------------------------
// Types for CDNA5 WMMA
// ---------------------------------------------------------------------------
typedef __attribute__((ext_vector_type(16))) __bf16 v16bf;
typedef __attribute__((ext_vector_type(8)))  __bf16 v8bf;
typedef __attribute__((ext_vector_type(8)))  float  v8f;

#define B_DIM   32
#define HW_DIM  1024
#define C_DIM   1024
#define K_DIM   1024
#define NROWS   (B_DIM * HW_DIM)          // 32768
#define INV_TT  (1.0f / 0.06f)
#define INV_PT  (1.0f / 0.12f)
#define EXP_MAX_C 50.0f
#define EPS_SK  1e-8f
#define EPS_NRM 1e-7f

// ---------------------------------------------------------------------------
// Wave32 / block reductions
// ---------------------------------------------------------------------------
__device__ __forceinline__ float waveReduceSum(float v) {
  #pragma unroll
  for (int off = 16; off > 0; off >>= 1) v += __shfl_xor(v, off, 32);
  return v;
}
__device__ __forceinline__ float waveReduceMax(float v) {
  #pragma unroll
  for (int off = 16; off > 0; off >>= 1) v = fmaxf(v, __shfl_xor(v, off, 32));
  return v;
}
__device__ __forceinline__ float blockReduceSum256(float v, float* sh) {
  const int lane = threadIdx.x & 31, wv = threadIdx.x >> 5;
  v = waveReduceSum(v);
  __syncthreads();
  if (lane == 0) sh[wv] = v;
  __syncthreads();
  float t = (threadIdx.x < 8) ? sh[threadIdx.x] : 0.0f;
  if (wv == 0) { t = waveReduceSum(t); if (lane == 0) sh[0] = t; }
  __syncthreads();
  return sh[0];
}
__device__ __forceinline__ float blockReduceMax256(float v, float* sh) {
  const int lane = threadIdx.x & 31, wv = threadIdx.x >> 5;
  v = waveReduceMax(v);
  __syncthreads();
  if (lane == 0) sh[wv] = v;
  __syncthreads();
  float t = (threadIdx.x < 8) ? sh[threadIdx.x] : -3.4e38f;
  if (wv == 0) { t = waveReduceMax(t); if (lane == 0) sh[0] = t; }
  __syncthreads();
  return sh[0];
}

// ---------------------------------------------------------------------------
// 1) L2 row-normalize x and split into bf16 hi/lo  (one block per row)
// ---------------------------------------------------------------------------
__global__ __launch_bounds__(256) void k_norm_split(const float* __restrict__ x,
                                                    __bf16* __restrict__ xhi,
                                                    __bf16* __restrict__ xlo) {
  __shared__ float sh[8];
  const size_t row = blockIdx.x;
  const float* xr = x + row * C_DIM;
  float v[4];
  float ss = 0.0f;
  #pragma unroll
  for (int j = 0; j < 4; ++j) {
    v[j] = xr[threadIdx.x + 256 * j];
    ss += v[j] * v[j];
  }
  const float tot = blockReduceSum256(ss, sh);
  const float inv = 1.0f / fmaxf(sqrtf(tot), EPS_NRM);
  #pragma unroll
  for (int j = 0; j < 4; ++j) {
    const float f = v[j] * inv;
    const __bf16 hi = (__bf16)f;
    const __bf16 lo = (__bf16)(f - (float)hi);
    xhi[row * C_DIM + threadIdx.x + 256 * j] = hi;
    xlo[row * C_DIM + threadIdx.x + 256 * j] = lo;
  }
}

// ---------------------------------------------------------------------------
// 2) Split W into bf16 hi/lo
// ---------------------------------------------------------------------------
__global__ __launch_bounds__(256) void k_w_split(const float* __restrict__ W,
                                                 __bf16* __restrict__ whi,
                                                 __bf16* __restrict__ wlo) {
  const size_t base = (size_t)blockIdx.x * 1024;
  #pragma unroll
  for (int j = 0; j < 4; ++j) {
    const size_t i = base + threadIdx.x + 256 * j;
    const float f = W[i];
    const __bf16 hi = (__bf16)f;
    const __bf16 lo = (__bf16)(f - (float)hi);
    whi[i] = hi;
    wlo[i] = lo;
  }
}

// ---------------------------------------------------------------------------
// 3) bf16x3 split-precision GEMM via v_wmma_f32_16x16x32_bf16
//    logits[b,n,k] = sum_c xn[b,n,c] * W[k,c]
//    Block: 128 threads (4 waves, 2x2) -> 128x128 tile; wave tile 64x64.
//    A-frag (16x32 bf16): lane lm=lane&15 is row M; h=lane>>4 selects
//    16 contiguous K at 16*h (matching B).
//    B-frag (32x16 bf16): lane lm is column N; 16 contiguous K at 16*h.
// ---------------------------------------------------------------------------
__device__ __forceinline__ v16bf load_a_frag(const __bf16* base, int h) {
  union { v16bf v; v8bf c[2]; } u;
  u.c[0] = *(const v8bf*)(base + 16 * h);
  u.c[1] = *(const v8bf*)(base + 16 * h + 8);
  return u.v;
}

__global__ __launch_bounds__(128) void k_gemm(const __bf16* __restrict__ Xhi,
                                              const __bf16* __restrict__ Xlo,
                                              const __bf16* __restrict__ Whi,
                                              const __bf16* __restrict__ Wlo,
                                              float* __restrict__ logits) {
  const int lane = threadIdx.x & 31;
  const int wave = threadIdx.x >> 5;
  const int wm = wave >> 1, wn = wave & 1;
  const int b  = blockIdx.z;
  const int m0 = blockIdx.x * 128 + wm * 64;
  const int n0 = blockIdx.y * 128 + wn * 64;
  const int lm = lane & 15;
  const int h  = lane >> 4;

  const __bf16* xh = Xhi + ((size_t)b << 20);
  const __bf16* xl = Xlo + ((size_t)b << 20);

  v8f acc[4][4];
  const v8f zero = {};
  #pragma unroll
  for (int i = 0; i < 4; ++i)
    #pragma unroll
    for (int j = 0; j < 4; ++j) acc[i][j] = zero;

  for (int kk = 0; kk < C_DIM; kk += 32) {
    v16bf ah[4], al[4];
    #pragma unroll
    for (int mt = 0; mt < 4; ++mt) {
      const size_t rb = (size_t)(m0 + mt * 16 + lm) * C_DIM + kk;
      ah[mt] = load_a_frag(xh + rb, h);
      al[mt] = load_a_frag(xl + rb, h);
    }
    #pragma unroll
    for (int nt = 0; nt < 4; ++nt) {
      const size_t cb = (size_t)(n0 + nt * 16 + lm) * C_DIM + kk + 16 * h;
      const v16bf bh = *(const v16bf*)(Whi + cb);
      const v16bf bl = *(const v16bf*)(Wlo + cb);
      #pragma unroll
      for (int mt = 0; mt < 4; ++mt) {
        acc[mt][nt] = __builtin_amdgcn_wmma_f32_16x16x32_bf16(
            false, ah[mt], false, bh, (short)0, acc[mt][nt], false, false);
        acc[mt][nt] = __builtin_amdgcn_wmma_f32_16x16x32_bf16(
            false, ah[mt], false, bl, (short)0, acc[mt][nt], false, false);
        acc[mt][nt] = __builtin_amdgcn_wmma_f32_16x16x32_bf16(
            false, al[mt], false, bh, (short)0, acc[mt][nt], false, false);
      }
    }
  }

  // C/D layout: lanes 0-15 -> N=lane, M=i ; lanes 16-31 -> N=lane-16, M=8+i
  float* outb = logits + ((size_t)b << 20);
  #pragma unroll
  for (int mt = 0; mt < 4; ++mt)
    #pragma unroll
    for (int nt = 0; nt < 4; ++nt) {
      const int col = n0 + nt * 16 + lm;
      #pragma unroll
      for (int i = 0; i < 8; ++i) {
        const int r = m0 + mt * 16 + h * 8 + i;
        outb[(size_t)r * K_DIM + col] = acc[mt][nt][i];
      }
    }
}

// ---------------------------------------------------------------------------
// 4) Column max of logits/0.06 (reduce over hw rows)
// ---------------------------------------------------------------------------
__global__ __launch_bounds__(256) void k_colmax(const float* __restrict__ logits,
                                                float* __restrict__ colstat) {
  const int col = blockIdx.x * 256 + threadIdx.x;  // 0..1023
  const int b = blockIdx.y;
  const float* p = logits + ((size_t)b << 20) + col;
  float m = -3.4e38f;
  for (int r = 0; r < HW_DIM; ++r) m = fmaxf(m, p[(size_t)r * K_DIM] * INV_TT);
  colstat[b * K_DIM + col] = m;
}

// 5) stable exp: A = exp(logit/TT + 50 - colmax)
__global__ __launch_bounds__(256) void k_exp(const float* __restrict__ logits,
                                             const float* __restrict__ colstat,
                                             float* __restrict__ A) {
  const int b = blockIdx.y;
  const size_t i = (size_t)blockIdx.x * 256 + threadIdx.x;  // within-batch idx
  const int col = (int)(i & 1023);
  const size_t gi = ((size_t)b << 20) + i;
  A[gi] = expf(logits[gi] * INV_TT + EXP_MAX_C - colstat[b * K_DIM + col]);
}

// 6) Column sums of A
__global__ __launch_bounds__(256) void k_colsum(const float* __restrict__ A,
                                                float* __restrict__ colstat) {
  const int col = blockIdx.x * 256 + threadIdx.x;
  const int b = blockIdx.y;
  const float* p = A + ((size_t)b << 20) + col;
  float s = 0.0f;
  for (int r = 0; r < HW_DIM; ++r) s += p[(size_t)r * K_DIM];
  colstat[b * K_DIM + col] = s;
}

// 7) Fused: col-divide + row-normalize, in place (one block per row)
__global__ __launch_bounds__(256) void k_rownorm(float* __restrict__ A,
                                                 const float* __restrict__ colstat) {
  __shared__ float sh[8];
  const size_t row = blockIdx.x;           // 0..32767
  const int b = (int)(row >> 10);
  float* ar = A + row * K_DIM;
  const float* cs = colstat + (size_t)b * K_DIM;
  float t[4];
  float s = 0.0f;
  #pragma unroll
  for (int j = 0; j < 4; ++j) {
    const int c = threadIdx.x + 256 * j;
    t[j] = ar[c] / (cs[c] + EPS_SK);
    s += t[j];
  }
  const float rs = blockReduceSum256(s, sh);
  const float inv = 1.0f / (rs + EPS_SK);
  #pragma unroll
  for (int j = 0; j < 4; ++j) ar[threadIdx.x + 256 * j] = t[j] * inv;
}

// ---------------------------------------------------------------------------
// 8) Per-row loss: rowloss = sum_k tgt*(p - lse(p)), p = logits/0.12
// ---------------------------------------------------------------------------
__global__ __launch_bounds__(256) void k_lossrow(const float* __restrict__ logits,
                                                 const float* __restrict__ A,
                                                 float* __restrict__ rowloss) {
  __shared__ float sh[8];
  const size_t row = blockIdx.x;
  const float* pr = logits + row * K_DIM;
  const float* tr = A + row * K_DIM;
  float pv[4], tv[4];
  float mx = -3.4e38f;
  #pragma unroll
  for (int j = 0; j < 4; ++j) {
    const int c = threadIdx.x + 256 * j;
    pv[j] = pr[c] * INV_PT;
    tv[j] = tr[c];
    mx = fmaxf(mx, pv[j]);
  }
  mx = blockReduceMax256(mx, sh);
  float se = 0.0f, dot = 0.0f, ts = 0.0f;
  #pragma unroll
  for (int j = 0; j < 4; ++j) {
    se += expf(pv[j] - mx);
    dot += tv[j] * pv[j];
    ts += tv[j];
  }
  se  = blockReduceSum256(se, sh);
  dot = blockReduceSum256(dot, sh);
  ts  = blockReduceSum256(ts, sh);
  if (threadIdx.x == 0) {
    const float lse = mx + logf(se);
    rowloss[row] = dot - lse * ts;
  }
}

// 9) Final reduce: loss = -mean(rowloss)
__global__ __launch_bounds__(256) void k_lossfinal(const float* __restrict__ rowloss,
                                                   float* __restrict__ loss_out) {
  __shared__ float sh[8];
  float s = 0.0f;
  for (int i = threadIdx.x; i < NROWS; i += 256) s += rowloss[i];
  s = blockReduceSum256(s, sh);
  if (threadIdx.x == 0) loss_out[0] = -s / (float)NROWS;
}

// ---------------------------------------------------------------------------
// Launcher
// ---------------------------------------------------------------------------
extern "C" void kernel_launch(void* const* d_in, const int* in_sizes, int n_in,
                              void* d_out, int out_size, void* d_ws, size_t ws_size,
                              hipStream_t stream) {
  (void)in_sizes; (void)n_in; (void)out_size; (void)ws_size;
  const float* x = (const float*)d_in[0];   // (32,32,32,1024) fp32
  const float* W = (const float*)d_in[1];   // (1024,1024) fp32
  float* out = (float*)d_out;               // 32M assignments + 1 loss

  char* ws = (char*)d_ws;
  __bf16* xhi = (__bf16*)(ws);                                   // 64 MB
  __bf16* xlo = (__bf16*)(ws + (size_t)64 * 1024 * 1024);        // 64 MB
  __bf16* whi = (__bf16*)(ws + (size_t)128 * 1024 * 1024);       // 2 MB
  __bf16* wlo = (__bf16*)(ws + (size_t)130 * 1024 * 1024);       // 2 MB
  float* logits  = (float*)(ws + (size_t)132 * 1024 * 1024);     // 128 MB
  float* colstat = (float*)(ws + (size_t)260 * 1024 * 1024);     // 128 KB
  float* rowloss = colstat + B_DIM * K_DIM;                      // 128 KB

  float* assign = out;                      // 32*1024*1024 floats
  float* loss   = out + (size_t)NROWS * K_DIM;

  // 1) normalize + split x
  k_norm_split<<<NROWS, 256, 0, stream>>>(x, xhi, xlo);
  // 2) split W
  k_w_split<<<1024, 256, 0, stream>>>(W, whi, wlo);
  // 3) GEMM -> logits
  dim3 ggrid(HW_DIM / 128, K_DIM / 128, B_DIM);
  k_gemm<<<ggrid, 128, 0, stream>>>(xhi, xlo, whi, wlo, logits);
  // 4) stable exp into d_out
  k_colmax<<<dim3(4, B_DIM), 256, 0, stream>>>(logits, colstat);
  k_exp<<<dim3(4096, B_DIM), 256, 0, stream>>>(logits, colstat, assign);
  // 5) 3 Sinkhorn iterations in place
  for (int it = 0; it < 3; ++it) {
    k_colsum<<<dim3(4, B_DIM), 256, 0, stream>>>(assign, colstat);
    k_rownorm<<<NROWS, 256, 0, stream>>>(assign, colstat);
  }
  // 6) loss
  k_lossrow<<<NROWS, 256, 0, stream>>>(logits, assign, rowloss);
  k_lossfinal<<<1, 256, 0, stream>>>(rowloss, loss);
}
